// FlashMHA_10685878632994
// MI455X (gfx1250) — matrix operand order repaired and checked by reading, not compiled, not run on hardware
//
#include <hip/hip_runtime.h>
#include <hip/hip_bf16.h>

#define BDIM 128
#define SDIM 256
#define EDIM 1024
#define HDIM 16
#define DDIM 64

typedef __attribute__((ext_vector_type(16))) __bf16 v16bf;
typedef __attribute__((ext_vector_type(8)))  float  v8f;
typedef __attribute__((ext_vector_type(8)))  __bf16 bf16x8;

__device__ __forceinline__ unsigned short f2bf(float f) {
  union { float f; unsigned u; } x; x.f = f;
  return (unsigned short)((x.u + 0x7FFFu + ((x.u >> 16) & 1u)) >> 16);  // RNE
}

__device__ __forceinline__ v8f wmma_bf16(v16bf a, v16bf b, v8f c) {
  return __builtin_amdgcn_wmma_f32_16x16x32_bf16(false, a, false, b, (short)0, c,
                                                 false, false);
}

// CDNA5 async copy: 16B global -> LDS, tracked with ASYNCcnt (no VGPR round-trip).
__device__ __forceinline__ void async_b128(unsigned short* lds_dst,
                                           const unsigned short* g) {
  unsigned l = (unsigned)(unsigned long long)lds_dst;   // LDS addr = low 32 bits
  unsigned long long ga = (unsigned long long)g;
  asm volatile("global_load_async_to_lds_b128 %0, %1, off"
               :: "v"(l), "v"(ga) : "memory");
}
__device__ __forceinline__ void wait_async0() {
  asm volatile("s_wait_asynccnt 0x0" ::: "memory");
}
// Async loads complete in order: waiting <=4 releases the current tile's 4
// copies while the next tile's 4 copies remain in flight (software pipeline).
__device__ __forceinline__ void wait_async4() {
  asm volatile("s_wait_asynccnt 0x4" ::: "memory");
}

// A-matrix fragment (16x32 bf16, row-major source). Per ISA layout:
// lanes 0-15: row, K elems {kblk+0..7, kblk+16..23}
// lanes 16-31: row, K elems {kblk+8..15, kblk+24..31}
__device__ __forceinline__ v16bf load_a_frag(const __bf16* base, int ld, int row,
                                             int kblk, int lane) {
  const __bf16* p = base + (size_t)row * ld + kblk + ((lane >> 4) << 3);
  union { v16bf v; bf16x8 h[2]; } u;
  u.h[0] = *(const bf16x8*)(p);
  u.h[1] = *(const bf16x8*)(p + 16);
  return u.v;
}

// B-matrix fragment (32x16 bf16), source is B^T row-major (column n contiguous).
// lanes 0-15: col n, K {kblk+0..7, kblk+16..23} ; lanes 16-31: col n, K {kblk+8..15, kblk+24..31}
__device__ __forceinline__ v16bf load_b_frag(const __bf16* base, int ld, int col,
                                             int kblk, int lane) {
  const __bf16* p = base + (size_t)col * ld + kblk + ((lane >> 4) << 3);
  union { v16bf v; bf16x8 h[2]; } u;
  u.h[0] = *(const bf16x8*)(p);
  u.h[1] = *(const bf16x8*)(p + 16);
  return u.v;
}

// Stage one 64x64 bf16 tile of A and B into LDS (4 x b128 per thread).
__device__ __forceinline__ void stage_tiles(unsigned short* sA, unsigned short* sB,
                                            const unsigned short* Ag,
                                            const unsigned short* Bg, int k0,
                                            int cr, int cc) {
  async_b128(&sA[cr * 72 + cc],     Ag + (size_t)cr * EDIM + k0 + cc);
  async_b128(&sA[cr * 72 + cc + 8], Ag + (size_t)cr * EDIM + k0 + cc + 8);
  async_b128(&sB[cr * 72 + cc],     Bg + (size_t)cr * EDIM + k0 + cc);
  async_b128(&sB[cr * 72 + cc + 8], Bg + (size_t)cr * EDIM + k0 + cc + 8);
}

__global__ __launch_bounds__(256) void cvt_kernel(const float* __restrict__ src,
                                                  unsigned short* __restrict__ dst,
                                                  int n) {
  int i = blockIdx.x * 256 + threadIdx.x;
  if (i < n) dst[i] = f2bf(src[i]);
}

// ---- QKV projection: C = X[b] @ W^T; M-tile 64, N-tile 64, K-tile 64 ----
// 8 waves (4x2); wave = 16 rows x 32 cols; double-buffered async LDS pipeline.
__global__ __launch_bounds__(256) void qkv_gemm_kernel(
    const __bf16* __restrict__ X, const __bf16* __restrict__ W,
    const float* __restrict__ bias, unsigned short* __restrict__ Qb,
    unsigned short* __restrict__ Kb, unsigned short* __restrict__ Vt) {
  __shared__ unsigned short sA[2][64 * 72];  // stride 72 elems = 144B (16B-aligned)
  __shared__ unsigned short sB[2][64 * 72];
  const int nb = blockIdx.x, mb = blockIdx.y, b = blockIdx.z;
  const int tid = threadIdx.x, lane = tid & 31, wave = tid >> 5;
  const int wm = wave & 3, wn = wave >> 2;
  const unsigned short* Ag =
      (const unsigned short*)X + ((size_t)b * SDIM + (size_t)mb * 64) * EDIM;
  const unsigned short* Bg = (const unsigned short*)W + (size_t)nb * 64 * EDIM;
  const int cr = tid >> 2;            // staging row 0..63
  const int cc = (tid & 3) << 4;      // staging col 0,16,32,48
  v8f acc[2] = {};
  stage_tiles(sA[0], sB[0], Ag, Bg, 0, cr, cc);    // prologue fill
  for (int k0 = 0; k0 < EDIM; k0 += 64) {
    const int cur = (k0 >> 6) & 1;
    const bool more = (k0 + 64 < EDIM);
    if (more) stage_tiles(sA[1 - cur], sB[1 - cur], Ag, Bg, k0 + 64, cr, cc);
    if (k0 + 128 < EDIM && tid < 64) {
      __builtin_prefetch(Ag + (size_t)tid * EDIM + k0 + 128, 0, 1);
      __builtin_prefetch(Bg + (size_t)tid * EDIM + k0 + 128, 0, 1);
    }
    if (more) wait_async4(); else wait_async0();
    __syncthreads();
#pragma unroll
    for (int kb = 0; kb < 64; kb += 32) {
      v16bf af = load_a_frag((const __bf16*)sA[cur], 72, wm * 16 + (lane & 15), kb, lane);
      v16bf b0 = load_b_frag((const __bf16*)sB[cur], 72, wn * 32 + (lane & 15), kb, lane);
      v16bf b1 = load_b_frag((const __bf16*)sB[cur], 72, wn * 32 + 16 + (lane & 15), kb, lane);
      acc[0] = wmma_bf16(af, b0, acc[0]);
      acc[1] = wmma_bf16(af, b1, acc[1]);
    }
    __syncthreads();   // protects buf[cur] before it is refilled in iter k+1
  }
  const int mhalf = (lane >> 4) << 3;   // C-layout: hi lanes hold rows m+8
  const int ncol = lane & 15;
#pragma unroll
  for (int t = 0; t < 2; ++t) {
    const int nglob = nb * 64 + wn * 32 + t * 16 + ncol;
    const int g = nglob >> 10, h = (nglob >> 6) & 15, d = nglob & 63;
    const float bv = bias[nglob];
    const size_t bh = (size_t)b * HDIM + h;
#pragma unroll
    for (int i = 0; i < 8; ++i) {
      const int s = mb * 64 + wm * 16 + mhalf + i;
      float val = acc[t][i] + bv;
      if (g == 0)
        Qb[(bh * SDIM + s) * DDIM + d] = f2bf(val * 0.125f);  // pre-scale Q
      else if (g == 1)
        Kb[(bh * SDIM + s) * DDIM + d] = f2bf(val);
      else
        Vt[(bh * DDIM + d) * SDIM + s] = f2bf(val);           // V transposed [d,s]
    }
  }
}

// ---- Attention: block per (b, h, 64 q-rows); 4 waves x (16 q x 256 k) ----
__global__ __launch_bounds__(128) void attn_kernel(
    const __bf16* __restrict__ Qb, const __bf16* __restrict__ Kb,
    const __bf16* __restrict__ Vt, const float* __restrict__ bias_table,
    const int* __restrict__ rel_index, unsigned short* __restrict__ ctx) {
  __shared__ unsigned short sAttn[4 * 16 * 264];  // per-wave 16x256, stride 264
  const int qt = blockIdx.x, h = blockIdx.y, b = blockIdx.z;
  const int tid = threadIdx.x, lane = tid & 31, wave = tid >> 5;
  const size_t bh = (size_t)b * HDIM + h;
  const __bf16* Q = Qb + bh * SDIM * DDIM;
  const __bf16* K = Kb + bh * SDIM * DDIM;
  const __bf16* V = Vt + bh * DDIM * SDIM;
  const int q0 = qt * 64 + wave * 16;
  const int mhalf = (lane >> 4) << 3;
  const int kcol = lane & 15;

  v16bf qf0 = load_a_frag(Q, DDIM, q0 + (lane & 15), 0, lane);
  v16bf qf1 = load_a_frag(Q, DDIM, q0 + (lane & 15), 32, lane);

  v8f acc[16] = {};
#pragma unroll
  for (int t = 0; t < 16; ++t) {
    v16bf b0 = load_b_frag(K, DDIM, t * 16 + (lane & 15), 0, lane);
    v16bf b1 = load_b_frag(K, DDIM, t * 16 + (lane & 15), 32, lane);
    acc[t] = wmma_bf16(qf0, b0, acc[t]);
    acc[t] = wmma_bf16(qf1, b1, acc[t]);
  }
#pragma unroll
  for (int t = 0; t < 16; ++t) {
#pragma unroll
    for (int i = 0; i < 8; ++i) {
      int qg = q0 + mhalf + i, kg = t * 16 + kcol;
      acc[t][i] += bias_table[rel_index[qg * SDIM + kg] * HDIM + h];
    }
  }
  // softmax over k (rows m<8 in lanes 0-15, m>=8 in lanes 16-31)
#pragma unroll
  for (int i = 0; i < 8; ++i) {
    float mx = -3.4e38f;
#pragma unroll
    for (int t = 0; t < 16; ++t) mx = fmaxf(mx, acc[t][i]);
#pragma unroll
    for (int m = 1; m < 16; m <<= 1) mx = fmaxf(mx, __shfl_xor(mx, m, 16));
    float sum = 0.f;
#pragma unroll
    for (int t = 0; t < 16; ++t) {
      float e = __expf(acc[t][i] - mx);
      acc[t][i] = e;
      sum += e;
    }
#pragma unroll
    for (int m = 1; m < 16; m <<= 1) sum += __shfl_xor(sum, m, 16);
    float inv = 1.f / sum;
#pragma unroll
    for (int t = 0; t < 16; ++t) acc[t][i] *= inv;
  }
  // restage attn (C layout -> A layout) through LDS as bf16
  unsigned short* myA = sAttn + wave * 16 * 264;
#pragma unroll
  for (int t = 0; t < 16; ++t)
#pragma unroll
    for (int i = 0; i < 8; ++i)
      myA[(mhalf + i) * 264 + t * 16 + kcol] = f2bf(acc[t][i]);
  __syncthreads();

  v8f cacc[4] = {};
  for (int kb = 0; kb < SDIM; kb += 32) {
    v16bf af = load_a_frag((const __bf16*)myA, 264, lane & 15, kb, lane);
#pragma unroll
    for (int dt = 0; dt < 4; ++dt) {
      v16bf bv = load_b_frag(V, SDIM, dt * 16 + (lane & 15), kb, lane);
      cacc[dt] = wmma_bf16(af, bv, cacc[dt]);
    }
  }
#pragma unroll
  for (int dt = 0; dt < 4; ++dt)
#pragma unroll
    for (int i = 0; i < 8; ++i) {
      int s = q0 + mhalf + i;
      int e = h * DDIM + dt * 16 + kcol;
      ctx[((size_t)b * SDIM + s) * EDIM + e] = f2bf(cacc[dt][i]);
    }
}

// ---- Output projection: out = ctx @ out_w^T + out_b (fp32 out) ----
__global__ __launch_bounds__(256) void out_gemm_kernel(
    const __bf16* __restrict__ Cx, const __bf16* __restrict__ W,
    const float* __restrict__ bias, float* __restrict__ out) {
  __shared__ unsigned short sA[2][64 * 72];
  __shared__ unsigned short sB[2][64 * 72];
  const int nb = blockIdx.x, mb = blockIdx.y;
  const int tid = threadIdx.x, lane = tid & 31, wave = tid >> 5;
  const int wm = wave & 3, wn = wave >> 2;
  const unsigned short* Ag = (const unsigned short*)Cx + (size_t)mb * 64 * EDIM;
  const unsigned short* Bg = (const unsigned short*)W + (size_t)nb * 64 * EDIM;
  const int cr = tid >> 2;
  const int cc = (tid & 3) << 4;
  v8f acc[2] = {};
  stage_tiles(sA[0], sB[0], Ag, Bg, 0, cr, cc);
  for (int k0 = 0; k0 < EDIM; k0 += 64) {
    const int cur = (k0 >> 6) & 1;
    const bool more = (k0 + 64 < EDIM);
    if (more) stage_tiles(sA[1 - cur], sB[1 - cur], Ag, Bg, k0 + 64, cr, cc);
    if (k0 + 128 < EDIM && tid < 64) {
      __builtin_prefetch(Ag + (size_t)tid * EDIM + k0 + 128, 0, 1);
      __builtin_prefetch(Bg + (size_t)tid * EDIM + k0 + 128, 0, 1);
    }
    if (more) wait_async4(); else wait_async0();
    __syncthreads();
#pragma unroll
    for (int kb = 0; kb < 64; kb += 32) {
      v16bf af = load_a_frag((const __bf16*)sA[cur], 72, wm * 16 + (lane & 15), kb, lane);
      v16bf b0 = load_b_frag((const __bf16*)sB[cur], 72, wn * 32 + (lane & 15), kb, lane);
      v16bf b1 = load_b_frag((const __bf16*)sB[cur], 72, wn * 32 + 16 + (lane & 15), kb, lane);
      acc[0] = wmma_bf16(af, b0, acc[0]);
      acc[1] = wmma_bf16(af, b1, acc[1]);
    }
    __syncthreads();
  }
  const int mhalf = (lane >> 4) << 3;
  const int ncol = lane & 15;
#pragma unroll
  for (int t = 0; t < 2; ++t) {
    const int nglob = nb * 64 + wn * 32 + t * 16 + ncol;
    const float bv = bias[nglob];
#pragma unroll
    for (int i = 0; i < 8; ++i) {
      const int row = mb * 64 + wm * 16 + mhalf + i;
      out[(size_t)row * EDIM + nglob] = acc[t][i] + bv;
    }
  }
}

extern "C" void kernel_launch(void* const* d_in, const int* in_sizes, int n_in,
                              void* d_out, int out_size, void* d_ws, size_t ws_size,
                              hipStream_t stream) {
  const float* x      = (const float*)d_in[0];
  const float* qkv_w  = (const float*)d_in[1];
  const float* qkv_b  = (const float*)d_in[2];
  const float* out_w  = (const float*)d_in[3];
  const float* out_b  = (const float*)d_in[4];
  const float* bias_t = (const float*)d_in[5];
  const int*   relidx = (const int*)d_in[6];
  float* out = (float*)d_out;
  (void)in_sizes; (void)n_in; (void)out_size; (void)ws_size;

  unsigned short* w = (unsigned short*)d_ws;
  const size_t nX  = (size_t)BDIM * SDIM * EDIM;
  const size_t nQW = (size_t)3 * EDIM * EDIM;
  const size_t nOW = (size_t)EDIM * EDIM;
  size_t off = 0;
  unsigned short* xbf   = w + off; off += nX;
  unsigned short* qkvwb = w + off; off += nQW;
  unsigned short* outwb = w + off; off += nOW;
  unsigned short* Qb    = w + off; off += nX;
  unsigned short* Kb    = w + off; off += nX;
  unsigned short* Vt    = w + off; off += nX;
  unsigned short* ctxb  = w + off; off += nX;

  cvt_kernel<<<dim3((unsigned)((nX  + 255) / 256)), 256, 0, stream>>>(x, xbf, (int)nX);
  cvt_kernel<<<dim3((unsigned)((nQW + 255) / 256)), 256, 0, stream>>>(qkv_w, qkvwb, (int)nQW);
  cvt_kernel<<<dim3((unsigned)((nOW + 255) / 256)), 256, 0, stream>>>(out_w, outwb, (int)nOW);

  qkv_gemm_kernel<<<dim3(48, 4, BDIM), 256, 0, stream>>>(
      (const __bf16*)xbf, (const __bf16*)qkvwb, qkv_b, Qb, Kb, Vt);

  attn_kernel<<<dim3(4, HDIM, BDIM), 128, 0, stream>>>(
      (const __bf16*)Qb, (const __bf16*)Kb, (const __bf16*)Vt, bias_t, relidx, ctxb);

  out_gemm_kernel<<<dim3(16, (BDIM * SDIM) / 64, 1), 256, 0, stream>>>(
      (const __bf16*)ctxb, (const __bf16*)outwb, out_b, out);
}
